// GEN1D_37048387895602
// MI455X (gfx1250) — matrix operand order repaired and checked by reading, not compiled, not run on hardware
//
#include <hip/hip_runtime.h>

// ---------------------------------------------------------------------------
// Chain-GCN on MI455X (gfx1250): split-f16 WMMA GEMM per step.
//   x[1024,256]; step: x = LN(x + A_norm @ (x@Wg) + bg)  (1023 steps, shared W)
// ---------------------------------------------------------------------------

#define N_NODES 1024
#define DF      256      // feature width
#define HF      128      // MLP hidden
#define PITCH   260      // LDS m-buffer row pitch (floats), padded vs 256
#define LN_EPS  1e-5f

typedef _Float16 v16h __attribute__((ext_vector_type(16)));
typedef _Float16 v8h  __attribute__((ext_vector_type(8)));
typedef float    v8f  __attribute__((ext_vector_type(8)));

// A-operand (16-bit 16x32, ISA 7.12.2): lane = hi*16 + m; half j holds
// K(j) = (j/8)*16 + hi*8 + (j%8)  -> two contiguous 8-half spans per lane.
__device__ __forceinline__ v16h load_a(const _Float16* __restrict__ rowp,
                                       int kc, int hi) {
  const v8h* p = (const v8h*)(rowp + kc * 32 + hi * 8);
  v8h lo = p[0];        // cols kc*32 + hi*8 + [0..7]
  v8h h2 = p[2];        // cols kc*32 + hi*8 + 16 + [0..7]
  return __builtin_shufflevector(lo, h2, 0,1,2,3,4,5,6,7,8,9,10,11,12,13,14,15);
}

// ---------------------------------------------------------------------------
// Pack Wg (f32 row-major [K=256, N=256]) into WMMA-B lane layout, split into
// hi/lo f16 halves.  B 16-bit 32x16 layout: lane = hi*16 + n, half j holds
// K = (j/8)*16 + hi*8 + (j%8) (same k order as the A operand).
// pack[((ct*8+kc)*32 + lane)*16 + j] ; lo copy at +65536.
// ---------------------------------------------------------------------------
__global__ void pack_wg_kernel(const float* __restrict__ Wg,
                               _Float16* __restrict__ pack) {
  int o    = blockIdx.x * 256 + threadIdx.x;   // 65536 elements
  int j    = o & 15;
  int lane = (o >> 4) & 31;
  int kc   = (o >> 9) & 7;
  int ct   = o >> 12;
  int hi   = lane >> 4;
  int n    = ct * 16 + (lane & 15);
  int k    = kc * 32 + (j >> 3) * 16 + hi * 8 + (j & 7);
  float v  = Wg[k * DF + n];
  _Float16 h = (_Float16)v;
  pack[o]         = h;
  pack[65536 + o] = (_Float16)(v - (float)h);
}

// ---------------------------------------------------------------------------
// Encoder: h = relu(X @ We1 + be1) @ We2 + be2 ; writes f32 + split-f16.
// One block (256 thr) per node.
// ---------------------------------------------------------------------------
__global__ __launch_bounds__(256) void encoder_kernel(
    const float* __restrict__ X,  const float* __restrict__ We1,
    const float* __restrict__ be1, const float* __restrict__ We2,
    const float* __restrict__ be2,
    float* __restrict__ h32, _Float16* __restrict__ hh, _Float16* __restrict__ hl) {
  __shared__ float h1[HF];
  int i = blockIdx.x, t = threadIdx.x;
  float xv = X[i];
  if (t < HF) h1[t] = fmaxf(xv * We1[t] + be1[t], 0.f);
  __syncthreads();
  float acc = be2[t];
  #pragma unroll 4
  for (int k = 0; k < HF; ++k) acc += h1[k] * We2[k * DF + t];
  h32[i * DF + t] = acc;
  _Float16 ah = (_Float16)acc;
  hh[i * DF + t] = ah;
  hl[i * DF + t] = (_Float16)(acc - (float)ah);
}

// ---------------------------------------------------------------------------
// One GCN step.  64 blocks x 256 threads (8 waves); block owns rows
// [R0, R0+16).  Wave w handles col-tiles {2w, 2w+1}; per tile: 8 K-chunks x 3
// split-f16 WMMAs.  Halo A-tile packs rows R0-1 / R0+16 into lanes 0/1.
// m -> LDS, then tridiagonal aggregate + residual + LayerNorm.
// ---------------------------------------------------------------------------
__global__ __launch_bounds__(256) void gcn_step_kernel(
    const float*    __restrict__ xs32,
    const _Float16* __restrict__ xsh, const _Float16* __restrict__ xsl,
    float*    __restrict__ xd32,
    _Float16* __restrict__ xdh, _Float16* __restrict__ xdl,
    const _Float16* __restrict__ pack,
    const float* __restrict__ bg, const float* __restrict__ gamma,
    const float* __restrict__ beta) {
  __shared__ float mlds[18 * PITCH];   // rows: 0 = R0-1, 1..16 = R0.., 17 = R0+16

  const int tid  = threadIdx.x;
  const int w    = tid >> 5;
  const int lane = tid & 31;
  const int m15  = lane & 15;
  const int hi   = lane >> 4;
  const int R0   = blockIdx.x << 4;

  const int mrow = R0 + m15;
  int hrow = R0;                                    // dummy rows -> weight 0 later
  if (m15 == 0)      hrow = (R0 > 0) ? (R0 - 1) : 0;
  else if (m15 == 1) hrow = (R0 + 16 < N_NODES) ? (R0 + 16) : (N_NODES - 1);

  const _Float16* rh = xsh + mrow * DF;
  const _Float16* rl = xsl + mrow * DF;
  const _Float16* qh = xsh + hrow * DF;
  const _Float16* ql = xsl + hrow * DF;

  #pragma unroll
  for (int t = 0; t < 2; ++t) {
    const int ct = (w << 1) + t;
    const v16h* ph = (const v16h*)pack + (ct * 8 * 32 + lane);
    const v16h* pl = ph + (65536 / 16);
    v16h Bh[8], Bl[8];
    #pragma unroll
    for (int kc = 0; kc < 8; ++kc) { Bh[kc] = ph[kc * 32]; Bl[kc] = pl[kc * 32]; }

    v8f c0 = {};                                    // main 16-row tile
    #pragma unroll
    for (int kc = 0; kc < 8; ++kc) {
      v16h ah = load_a(rh, kc, hi);
      v16h al = load_a(rl, kc, hi);
      c0 = __builtin_amdgcn_wmma_f32_16x16x32_f16(false, ah, false, Bh[kc], (short)0, c0, false, false);
      c0 = __builtin_amdgcn_wmma_f32_16x16x32_f16(false, ah, false, Bl[kc], (short)0, c0, false, false);
      c0 = __builtin_amdgcn_wmma_f32_16x16x32_f16(false, al, false, Bh[kc], (short)0, c0, false, false);
    }
    v8f c1 = {};                                    // halo tile (rows R0-1, R0+16)
    #pragma unroll
    for (int kc = 0; kc < 8; ++kc) {
      v16h ah = load_a(qh, kc, hi);
      v16h al = load_a(ql, kc, hi);
      c1 = __builtin_amdgcn_wmma_f32_16x16x32_f16(false, ah, false, Bh[kc], (short)0, c1, false, false);
      c1 = __builtin_amdgcn_wmma_f32_16x16x32_f16(false, ah, false, Bl[kc], (short)0, c1, false, false);
      c1 = __builtin_amdgcn_wmma_f32_16x16x32_f16(false, al, false, Bh[kc], (short)0, c1, false, false);
    }
    // C layout: VGPR r holds D[r + 8*hi][lane&15]
    const int col = (ct << 4) + m15;
    #pragma unroll
    for (int r = 0; r < 8; ++r)
      mlds[(1 + r + (hi << 3)) * PITCH + col] = c0[r];
    if (lane < 16) {                                // halo rows 0 and 1 of tile
      mlds[col]              = c1[0];
      mlds[17 * PITCH + col] = c1[1];
    }
  }
  __syncthreads();

  // ---- aggregate + residual + LayerNorm: 2 rows per wave, cols lane+32k ----
  #pragma unroll
  for (int rr = 0; rr < 2; ++rr) {
    const int rloc = (w << 1) + rr;
    const int i    = R0 + rloc;
    const int lr   = rloc + 1;
    const float dc = (i == 0 || i == N_NODES - 1) ? 2.f : 3.f;
    float wl = 0.f, wr = 0.f;
    if (i > 0)           { float dl = (i - 1 == 0) ? 2.f : 3.f;           wl = rsqrtf(dl * dc); }
    if (i < N_NODES - 1) { float dr = (i + 1 == N_NODES - 1) ? 2.f : 3.f; wr = rsqrtf(dc * dr); }
    const float wc = 1.f / dc;

    float y[8], s = 0.f;
    #pragma unroll
    for (int k = 0; k < 8; ++k) {
      int col = lane + (k << 5);
      float agg = wl * mlds[(lr - 1) * PITCH + col]
                + wc * mlds[lr * PITCH + col]
                + wr * mlds[(lr + 1) * PITCH + col] + bg[col];
      y[k] = xs32[i * DF + col] + agg;
      s += y[k];
    }
    #pragma unroll
    for (int o = 16; o > 0; o >>= 1) s += __shfl_xor(s, o);
    const float mu = s * (1.f / (float)DF);
    float v = 0.f;
    #pragma unroll
    for (int k = 0; k < 8; ++k) { float d = y[k] - mu; v += d * d; }
    #pragma unroll
    for (int o = 16; o > 0; o >>= 1) v += __shfl_xor(v, o);
    const float inv = rsqrtf(v * (1.f / (float)DF) + LN_EPS);
    #pragma unroll
    for (int k = 0; k < 8; ++k) {
      int col = lane + (k << 5);
      float o = gamma[col] * (y[k] - mu) * inv + beta[col];
      xd32[i * DF + col] = o;
      _Float16 oh = (_Float16)o;
      xdh[i * DF + col] = oh;
      xdl[i * DF + col] = (_Float16)(o - (float)oh);
    }
  }
}

// ---------------------------------------------------------------------------
// Decoder: out = relu(h @ Wd1 + bd1) @ Wd2 + bd2.  One block (128 thr)/node.
// ---------------------------------------------------------------------------
__global__ __launch_bounds__(128) void decoder_kernel(
    const float* __restrict__ h,  const float* __restrict__ Wd1,
    const float* __restrict__ bd1, const float* __restrict__ Wd2,
    const float* __restrict__ bd2, float* __restrict__ out) {
  __shared__ float red[HF];
  int i = blockIdx.x, t = threadIdx.x;
  float acc = bd1[t];
  #pragma unroll 4
  for (int c = 0; c < DF; ++c) acc += h[i * DF + c] * Wd1[c * HF + t];
  red[t] = fmaxf(acc, 0.f) * Wd2[t];
  __syncthreads();
  for (int sft = HF / 2; sft > 0; sft >>= 1) {
    if (t < sft) red[t] += red[t + sft];
    __syncthreads();
  }
  if (t == 0) out[i] = red[0] + bd2[0];
}

// ---------------------------------------------------------------------------
extern "C" void kernel_launch(void* const* d_in, const int* in_sizes, int n_in,
                              void* d_out, int out_size, void* d_ws, size_t ws_size,
                              hipStream_t stream) {
  (void)in_sizes; (void)n_in; (void)out_size; (void)ws_size;
  const float* X     = (const float*)d_in[0];
  const float* We1   = (const float*)d_in[1];
  const float* be1   = (const float*)d_in[2];
  const float* We2   = (const float*)d_in[3];
  const float* be2   = (const float*)d_in[4];
  const float* Wg    = (const float*)d_in[5];
  const float* bg    = (const float*)d_in[6];
  const float* gamma = (const float*)d_in[7];
  const float* beta  = (const float*)d_in[8];
  const float* Wd1   = (const float*)d_in[9];
  const float* bd1   = (const float*)d_in[10];
  const float* Wd2   = (const float*)d_in[11];
  const float* bd2   = (const float*)d_in[12];
  // src/dst (d_in[13..14]) encode the fixed chain graph; structure used directly.

  char* ws = (char*)d_ws;
  _Float16* pack = (_Float16*)(ws);                        // 256 KB (hi+lo)
  float*    x0   = (float*)   (ws + (256u << 10));         // 1 MB
  float*    x1   = (float*)   (ws + (256u << 10) + (1u << 20));
  _Float16* x0h  = (_Float16*)(ws + (256u << 10) + (2u << 20));
  _Float16* x1h  = (_Float16*)(ws + (256u << 10) + (2u << 20) + (512u << 10));
  _Float16* x0l  = (_Float16*)(ws + (256u << 10) + (3u << 20));
  _Float16* x1l  = (_Float16*)(ws + (256u << 10) + (3u << 20) + (512u << 10));

  pack_wg_kernel<<<256, 256, 0, stream>>>(Wg, pack);
  encoder_kernel<<<N_NODES, 256, 0, stream>>>(X, We1, be1, We2, be2, x0, x0h, x0l);

  for (int s = 0; s < N_NODES - 1; ++s) {
    if ((s & 1) == 0)
      gcn_step_kernel<<<64, 256, 0, stream>>>(x0, x0h, x0l, x1, x1h, x1l,
                                              pack, bg, gamma, beta);
    else
      gcn_step_kernel<<<64, 256, 0, stream>>>(x1, x1h, x1l, x0, x0h, x0l,
                                              pack, bg, gamma, beta);
  }
  // 1023 steps: final state lands in x1.
  decoder_kernel<<<N_NODES, 128, 0, stream>>>(x1, Wd1, bd1, Wd2, bd2, (float*)d_out);
}
